// TextBlock_70437463654799
// MI455X (gfx1250) — matrix operand order repaired and checked by reading, not compiled, not run on hardware
//
#include <hip/hip_runtime.h>

// ---------------------------------------------------------------------------
// Transformer block (LN1 -> per-head QKV -> causal attn (T=8) -> +res -> LN2
// -> FFN(relu) -> +idx) for MI455X (gfx1250).
// f16 WMMA (f32 accum) GEMMs with double-buffered LDS fed by
// GLOBAL_LOAD_ASYNC_TO_LDS_B128 (ASYNCcnt) to overlap HBM with matrix math.
// ---------------------------------------------------------------------------

typedef __attribute__((ext_vector_type(16))) _Float16 v16h;
typedef __attribute__((ext_vector_type(8)))  _Float16 v8h;
typedef __attribute__((ext_vector_type(8)))  float    v8f;

#define BDIM 8192
#define TSEQ 8
#define CDIM 512
#define HN   8
#define HSZ  64
#define FF   2048
#define MTOK (BDIM*TSEQ)   // 65536 tokens
#define NQKV (3*CDIM)      // 1536

// ---------------------------- weight packing -------------------------------
__global__ __launch_bounds__(256) void pack_qkv_w(
    const float* __restrict__ Wp, const float* __restrict__ Wk,
    const float* __restrict__ Wv, const float* __restrict__ bp,
    const float* __restrict__ bk, const float* __restrict__ bv,
    _Float16* __restrict__ Wt, float* __restrict__ bq)
{
  int i = blockIdx.x * 256 + threadIdx.x;
  if (i < NQKV * CDIM) {
    int n = i >> 9;            // 0..1535
    int c = i & (CDIM - 1);    // 0..511
    const float* W = (n < CDIM) ? Wp : ((n < 2 * CDIM) ? Wk : Wv);
    int nn = n & (CDIM - 1);
    int h = nn >> 6, d = nn & 63;
    Wt[i] = (_Float16)W[(h * CDIM + c) * HSZ + d];
  }
  if (i < NQKV) {
    const float* bs = (i < CDIM) ? bp : ((i < 2 * CDIM) ? bk : bv);
    int nn = i & (CDIM - 1);
    bq[i] = bs[(nn >> 6) * HSZ + (nn & 63)];
  }
}

__global__ __launch_bounds__(256) void pack_w1(const float* __restrict__ W1,
                                               _Float16* __restrict__ Wt)
{
  int i = blockIdx.x * 256 + threadIdx.x;
  if (i < FF * CDIM) {
    int n = i >> 9;
    int c = i & (CDIM - 1);
    Wt[i] = (_Float16)W1[c * FF + n];
  }
}

__global__ __launch_bounds__(256) void pack_w2(const float* __restrict__ W2,
                                               _Float16* __restrict__ Wt)
{
  int i = blockIdx.x * 256 + threadIdx.x;
  if (i < CDIM * FF) {
    int n = i >> 11;
    int f = i & (FF - 1);
    Wt[i] = (_Float16)W2[f * CDIM + n];
  }
}

// ------------------------------- LayerNorm ---------------------------------
__global__ __launch_bounds__(256) void ln_rows(const float* __restrict__ x,
    const float* __restrict__ g, const float* __restrict__ bb,
    _Float16* __restrict__ out)
{
  int wid = threadIdx.x >> 5, lane = threadIdx.x & 31;
  size_t row = (size_t)blockIdx.x * 8 + wid;
  const float* xr = x + row * CDIM;
  float vals[16], s = 0.f, ss = 0.f;
#pragma unroll
  for (int j = 0; j < 16; j++) {
    float v = xr[lane + 32 * j];
    vals[j] = v; s += v; ss += v * v;
  }
#pragma unroll
  for (int o = 16; o >= 1; o >>= 1) {
    s  += __shfl_xor(s,  o, 32);
    ss += __shfl_xor(ss, o, 32);
  }
  float mu = s * (1.f / CDIM);
  float var = ss * (1.f / CDIM) - mu * mu;
  float rstd = rsqrtf(var + 1e-5f);
  _Float16* orow = out + row * CDIM;
#pragma unroll
  for (int j = 0; j < 16; j++) {
    int c = lane + 32 * j;
    orow[c] = (_Float16)((vals[j] - mu) * rstd * g[c] + bb[c]);
  }
}

// --------------------------- async copy helpers ----------------------------
// Flat LDS addresses keep the wave-relative LDS offset in addr[31:0], so a
// 32-bit truncation of a generic pointer to __shared__ is the VDST operand.
__device__ __forceinline__ unsigned lds_off(const void* p)
{
  return (unsigned)(size_t)p;
}
// 16B per lane, memory -> LDS, tracked by ASYNCcnt.
__device__ __forceinline__ void async_b128(unsigned lds, const _Float16* g)
{
  asm volatile("global_load_async_to_lds_b128 %0, %1, off"
               :: "v"(lds), "v"(g) : "memory");
}

// --------------------------- WMMA fragment loads ---------------------------
// A 16x32 f16 (ISA 7.12.2): lane<16 -> M=lane, K 0..7 & 16..23; lane>=16 ->
// K 8..15 & 24..31 : two contiguous 128b LDS loads.
__device__ __forceinline__ v16h frag_a(const _Float16* s, int m0, int ld)
{
  int lane = threadIdx.x & 31;
  const _Float16* p = s + (m0 + (lane & 15)) * ld + ((lane >> 4) << 3);
  v8h lo = *(const v8h*)p;
  v8h hi = *(const v8h*)(p + 16);
  return __builtin_shufflevector(lo, hi, 0,1,2,3,4,5,6,7,8,9,10,11,12,13,14,15);
}
// B 32x16 f16 from [N][K] LDS: lane<16 -> N=lane, K 0..7 & 16..23; lane>=16 -> K 8..15 & 24..31.
__device__ __forceinline__ v16h frag_b(const _Float16* s, int n0, int ld)
{
  int lane = threadIdx.x & 31;
  const _Float16* p = s + (n0 + (lane & 15)) * ld + ((lane >> 4) << 3);
  v8h lo = *(const v8h*)p;
  v8h hi = *(const v8h*)(p + 16);
  return __builtin_shufflevector(lo, hi, 0,1,2,3,4,5,6,7,8,9,10,11,12,13,14,15);
}

// ------------------------------- WMMA GEMM ---------------------------------
// C[M,N] = A[M,K] * Bt[N,K]^T + bias. Block tile 128x256, 8 waves (2x4),
// wave tile 64x64 = 4x4 wmma tiles (16 wmma : 16 ds_load_b128 per K-step).
// Double-buffered LDS filled by async global->LDS; K-step 32.
// OUTMODE: 0 = f16 out; 1 = f16 out + ReLU; 2 = f32 out + residual (FFN2).
template<int OUTMODE>
__global__ __launch_bounds__(256) void gemm_wmma(
    const _Float16* __restrict__ A, const _Float16* __restrict__ Bt,
    const float* __restrict__ bias, const float* __restrict__ res,
    void* __restrict__ out, int N, int K)
{
  constexpr int LD = 40;                 // 32 + 8 pad halves (80B rows)
  __shared__ _Float16 sA[2][128 * LD];   // 20 KB
  __shared__ _Float16 sB[2][256 * LD];   // 40 KB
  const int tid = threadIdx.x;
  const int bn = blockIdx.x * 256, bm = blockIdx.y * 128;
  const int wid = tid >> 5;
  const int wm = (wid & 1) * 64, wn = (wid >> 1) * 64;

  v8f zero = {};
  v8f acc[4][4];
#pragma unroll
  for (int i = 0; i < 4; i++)
#pragma unroll
    for (int j = 0; j < 4; j++) acc[i][j] = zero;

  // per-thread copy slots: 6 x 16B chunks (2 for A rows, 4 for B rows)
  const int crow = tid >> 2;        // 0..63
  const int cch  = (tid & 3) * 8;   // half offset within 32-wide K slab
  const _Float16* gA = A  + (size_t)(bm + crow) * K + cch;
  const _Float16* gB = Bt + (size_t)(bn + crow) * K + cch;
  const int KT = K / 32;

  auto issue = [&](int kt, int b) {
    const _Float16* ga = gA + kt * 32;
    async_b128(lds_off(&sA[b][crow * LD + cch]),        ga);
    async_b128(lds_off(&sA[b][(crow + 64) * LD + cch]), ga + (size_t)64 * K);
    const _Float16* gb = gB + kt * 32;
#pragma unroll
    for (int s = 0; s < 4; s++)
      async_b128(lds_off(&sB[b][(crow + s * 64) * LD + cch]),
                 gb + (size_t)(s * 64) * K);
  };

  issue(0, 0);
  for (int kt = 0; kt < KT; ++kt) {
    __syncthreads();                       // buf[(kt+1)&1] reads (kt-1) done
    if (kt + 1 < KT) {
      issue(kt + 1, (kt + 1) & 1);         // prefetch next tile
      asm volatile("s_wait_asynccnt 6" ::: "memory");  // tile kt landed
    } else {
      asm volatile("s_wait_asynccnt 0" ::: "memory");
    }
    __syncthreads();                       // all waves' tile kt in LDS

    const int cur = kt & 1;
    v16h af[4];
#pragma unroll
    for (int i = 0; i < 4; i++) af[i] = frag_a(sA[cur], wm + i * 16, LD);
#pragma unroll
    for (int j = 0; j < 4; j++) {
      v16h bf = frag_b(sB[cur], wn + j * 16, LD);
#pragma unroll
      for (int i = 0; i < 4; i++)
        acc[i][j] = __builtin_amdgcn_wmma_f32_16x16x32_f16(
            false, af[i], false, bf, (short)0, acc[i][j], false, false);
    }
  }

  // Epilogue: C/D layout -> vgpr r holds M = r + 8*(lane>=16), N = lane&15.
  const int lane = tid & 31;
  const int cn = lane & 15;
  const int rh = (lane >> 4) * 8;
#pragma unroll
  for (int i = 0; i < 4; i++) {
    int gm0 = bm + wm + i * 16 + rh;
#pragma unroll
    for (int j = 0; j < 4; j++) {
      int gn = bn + wn + j * 16 + cn;
      float bv = bias[gn];
#pragma unroll
      for (int r = 0; r < 8; r++) {
        size_t oidx = (size_t)(gm0 + r) * N + gn;
        float v = acc[i][j][r] + bv;
        if (OUTMODE == 1) v = fmaxf(v, 0.f);
        if (OUTMODE == 2) ((float*)out)[oidx] = v + res[oidx];
        else              ((_Float16*)out)[oidx] = (_Float16)v;
      }
    }
  }
}

// ------------------- fused attention + residual + LN2 ----------------------
__global__ __launch_bounds__(256) void attn_ln2(
    const _Float16* __restrict__ qkv, const float* __restrict__ idx,
    const float* __restrict__ g2, const float* __restrict__ bb2,
    _Float16* __restrict__ hln2)
{
  __shared__ float sx[TSEQ * NQKV];      // [t][q|k|v] as f32  (48 KB)
  __shared__ float sw[HN * TSEQ * TSEQ]; // wei [h][t][s]       (2 KB)
  const int b = blockIdx.x;
  const int tid = threadIdx.x;

  const _Float16* src = qkv + (size_t)b * TSEQ * NQKV;
  for (int c = tid; c < TSEQ * NQKV / 8; c += 256) {
    v8h v = *(const v8h*)(src + c * 8);
#pragma unroll
    for (int e = 0; e < 8; e++) sx[c * 8 + e] = (float)v[e];
  }
  __syncthreads();

  const float scale = 0.125f;  // 1/sqrt(64)
  for (int i = tid; i < HN * TSEQ * TSEQ; i += 256) {
    int h = i >> 6, t = (i >> 3) & 7, s = i & 7;
    const float* qp = sx + t * NQKV + h * HSZ;
    const float* kp = sx + s * NQKV + CDIM + h * HSZ;
    float d = 0.f;
#pragma unroll
    for (int c = 0; c < HSZ; c++) d += qp[c] * kp[c];
    sw[i] = d * scale;
  }
  __syncthreads();

  if (tid < HN * TSEQ) {
    int t = tid & 7;
    float* w = sw + tid * TSEQ;
    float mx = -3.4e38f;
    for (int s = 0; s <= t; s++) mx = fmaxf(mx, w[s]);
    float sum = 0.f;
#pragma unroll
    for (int s = 0; s < TSEQ; s++) {
      float e = (s <= t) ? __expf(w[s] - mx) : 0.f;
      w[s] = e; sum += e;
    }
    float inv = 1.f / sum;
#pragma unroll
    for (int s = 0; s < TSEQ; s++) w[s] *= inv;
  }
  __syncthreads();

  // att + idx -> reuse q slot of sx (q dead after wei stage)
  for (int i = tid; i < TSEQ * CDIM; i += 256) {
    int t = i >> 9, c = i & (CDIM - 1);
    int h = c >> 6, d = c & 63;
    const float* w = sw + (h * TSEQ + t) * TSEQ;
    float a = 0.f;
#pragma unroll
    for (int s = 0; s < TSEQ; s++)
      a += w[s] * sx[s * NQKV + 2 * CDIM + h * HSZ + d];
    sx[t * NQKV + c] = a + idx[((size_t)b * TSEQ + t) * CDIM + c];
  }
  __syncthreads();

  int wid = tid >> 5, lane = tid & 31, t = wid;
  float vals[16], s = 0.f, ss = 0.f;
#pragma unroll
  for (int j = 0; j < 16; j++) {
    float v = sx[t * NQKV + lane + 32 * j];
    vals[j] = v; s += v; ss += v * v;
  }
#pragma unroll
  for (int o = 16; o >= 1; o >>= 1) {
    s  += __shfl_xor(s,  o, 32);
    ss += __shfl_xor(ss, o, 32);
  }
  float mu = s * (1.f / CDIM);
  float var = ss * (1.f / CDIM) - mu * mu;
  float rstd = rsqrtf(var + 1e-5f);
  _Float16* orow = hln2 + ((size_t)b * TSEQ + t) * CDIM;
#pragma unroll
  for (int j = 0; j < 16; j++) {
    int c = lane + 32 * j;
    orow[c] = (_Float16)((vals[j] - mu) * rstd * g2[c] + bb2[c]);
  }
}

// ------------------------------- launcher ----------------------------------
extern "C" void kernel_launch(void* const* d_in, const int* in_sizes, int n_in,
                              void* d_out, int out_size, void* d_ws, size_t ws_size,
                              hipStream_t stream)
{
  (void)in_sizes; (void)n_in; (void)out_size; (void)ws_size;
  const float* idx   = (const float*)d_in[0];
  const float* ln1_g = (const float*)d_in[1];
  const float* ln1_b = (const float*)d_in[2];
  const float* ln2_g = (const float*)d_in[3];
  const float* ln2_b = (const float*)d_in[4];
  const float* Wp    = (const float*)d_in[5];
  const float* bp    = (const float*)d_in[6];
  const float* Wk    = (const float*)d_in[7];
  const float* bk    = (const float*)d_in[8];
  const float* Wv    = (const float*)d_in[9];
  const float* bv    = (const float*)d_in[10];
  const float* W1    = (const float*)d_in[11];
  const float* b1    = (const float*)d_in[12];
  const float* W2    = (const float*)d_in[13];
  const float* b2    = (const float*)d_in[14];

  char* ws = (char*)d_ws;
  size_t off = 0;
  auto carve = [&](size_t bytes) {
    char* p = ws + off;
    off += (bytes + 255) & ~(size_t)255;
    return p;
  };
  _Float16* WqkvT = (_Float16*)carve((size_t)NQKV * CDIM * 2);
  float*    bqkv  = (float*)   carve((size_t)NQKV * 4);
  _Float16* W1T   = (_Float16*)carve((size_t)FF * CDIM * 2);
  _Float16* W2T   = (_Float16*)carve((size_t)CDIM * FF * 2);
  _Float16* xln   = (_Float16*)carve((size_t)MTOK * CDIM * 2);
  _Float16* qkvb  = (_Float16*)carve((size_t)MTOK * NQKV * 2);
  _Float16* hln2  = (_Float16*)carve((size_t)MTOK * CDIM * 2);
  _Float16* ffn1  = (_Float16*)carve((size_t)MTOK * FF * 2);

  pack_qkv_w<<<(NQKV * CDIM + 255) / 256, 256, 0, stream>>>(
      Wp, Wk, Wv, bp, bk, bv, WqkvT, bqkv);
  pack_w1<<<(FF * CDIM + 255) / 256, 256, 0, stream>>>(W1, W1T);
  pack_w2<<<(CDIM * FF + 255) / 256, 256, 0, stream>>>(W2, W2T);

  ln_rows<<<MTOK / 8, 256, 0, stream>>>(idx, ln1_g, ln1_b, xln);

  gemm_wmma<0><<<dim3(NQKV / 256, MTOK / 128), 256, 0, stream>>>(
      xln, WqkvT, bqkv, nullptr, qkvb, NQKV, CDIM);

  attn_ln2<<<BDIM, 256, 0, stream>>>(qkvb, idx, ln2_g, ln2_b, hln2);

  gemm_wmma<1><<<dim3(FF / 256, MTOK / 128), 256, 0, stream>>>(
      hln2, W1T, b1, nullptr, ffn1, FF, CDIM);

  gemm_wmma<2><<<dim3(CDIM / 256, MTOK / 128), 256, 0, stream>>>(
      ffn1, W2T, b2, idx, d_out, CDIM, FF);
}
